// CrossAttention_59322088292867
// MI455X (gfx1250) — matrix operand order repaired and checked by reading, not compiled, not run on hardware
//
#include <hip/hip_runtime.h>

// ---------------------------------------------------------------------------
// CDNA5 (gfx1250) cross-attention: all GEMMs via v_wmma_f32_16x16x32_bf16.
// wave32. Strides are compile-time so all fragment addressing lowers to
// SADDR(SGPR base) + one loop-invariant 32-bit voffset + instruction immediates.
// __launch_bounds__(32,1) => full VGPR budget, no occupancy-driven spills.
// ---------------------------------------------------------------------------

typedef __attribute__((ext_vector_type(16))) __bf16 v16bf;
typedef __attribute__((ext_vector_type(8)))  __bf16 v8bf;
typedef __attribute__((ext_vector_type(8)))  float  v8f;

#define DEV static __device__ __forceinline__

// A-fragment from a fully formed per-lane pointer (lane offset already added):
// loads K = {0..7} at p, K = {8..15} at p+8 (each 16B, contiguous).
DEV v16bf frag_a_at(const __bf16* p) {
  v8bf lo = *(const v8bf*)(p);
  v8bf hi = *(const v8bf*)(p + 8);
  v16bf a;
#pragma unroll
  for (int e = 0; e < 8; ++e) { a[e] = lo[e]; a[e + 8] = hi[e]; }
  return a;
}

// B-fragment: 32 contiguous bytes at the per-lane pointer.
DEV v16bf frag_b_at(const __bf16* p) { return *(const v16bf*)(p); }

DEV v8f wmma_bf16(v16bf a, v16bf b, v8f c) {
  return __builtin_amdgcn_wmma_f32_16x16x32_bf16(false, a, false, b, (short)0, c,
                                                 false, false);
}

// ---------------------------------------------------------------------------
// Conversion kernels
// ---------------------------------------------------------------------------
__global__ void cvt_f32_bf16(const float* __restrict__ s, __bf16* __restrict__ d,
                             int n) {
  int i = blockIdx.x * blockDim.x + threadIdx.x;
  if (i < n) d[i] = (__bf16)s[i];
}

// src [R][C] f32 -> dst [C][R] bf16
__global__ void cvt_f32_bf16_T(const float* __restrict__ s, __bf16* __restrict__ d,
                               int R, int C) {
  int c = blockIdx.x * blockDim.x + threadIdx.x;
  int r = blockIdx.y;
  if (c < C) d[(size_t)c * R + r] = (__bf16)s[(size_t)r * C + c];
}

// ---------------------------------------------------------------------------
// bf16 WMMA GEMM: C[M,N] = scale * (A[M,K] @ Bt[N,K]^T) (+ bias). K constexpr.
// One wave: 64(M) x 64(N) tile = 16 C-frags; 16 WMMAs per 16 b128 loads.
// k-loop advances the uniform base pointers (SALU); lane offsets are invariant.
// ---------------------------------------------------------------------------
enum { OUT_BF16 = 0, OUT_BF16_T = 1, OUT_F32_BIAS = 2 };

template <int MODE, int K>
__global__ __launch_bounds__(32, 1) void gemm_bf16_wmma(
    const __bf16* __restrict__ A, const __bf16* __restrict__ Bt,
    void* __restrict__ Cout, const float* __restrict__ bias,
    int M, int N, float scale) {
  const int n0 = blockIdx.x * 64;
  const int m0 = blockIdx.y * 64;
  const unsigned lane = threadIdx.x & 31u;
  const unsigned g = lane >> 4, nl = lane & 15;

  // Loop-invariant per-lane element offsets (32-bit).
  const unsigned offA = (unsigned)(m0 + (int)nl) * K + g * 16;
  const unsigned offB = (unsigned)(n0 + (int)nl) * K + g * 16;

  v8f acc[4][4];
#pragma unroll
  for (int mi = 0; mi < 4; ++mi)
#pragma unroll
    for (int nj = 0; nj < 4; ++nj) acc[mi][nj] = (v8f){};

  for (int k0 = 0; k0 < K; k0 += 32) {
    const __bf16* Ak = A + k0;   // uniform: SALU pointer advance
    const __bf16* Bk = Bt + k0;
    v16bf a[4];
#pragma unroll
    for (int mi = 0; mi < 4; ++mi)
      a[mi] = frag_a_at(Ak + offA + mi * 16 * K);  // const delta -> immediate
#pragma unroll
    for (int nj = 0; nj < 4; ++nj) {
      v16bf b = frag_b_at(Bk + offB + nj * 16 * K);
#pragma unroll
      for (int mi = 0; mi < 4; ++mi) acc[mi][nj] = wmma_bf16(a[mi], b, acc[mi][nj]);
    }
  }

#pragma unroll
  for (int mi = 0; mi < 4; ++mi) {
#pragma unroll
    for (int nj = 0; nj < 4; ++nj) {
#pragma unroll
      for (int r = 0; r < 8; ++r) {
        const int m = m0 + 16 * mi + r + 8 * (int)g;
        const int n = n0 + 16 * nj + (int)nl;
        float val = acc[mi][nj][r] * scale;
        if (MODE == OUT_BF16) {
          ((__bf16*)Cout)[(size_t)m * N + n] = (__bf16)val;
        } else if (MODE == OUT_BF16_T) {
          ((__bf16*)Cout)[(size_t)n * M + m] = (__bf16)val;
        } else {
          ((float*)Cout)[(size_t)m * N + n] = val + bias[n];
        }
      }
    }
  }
}

// ---------------------------------------------------------------------------
// Flash-attention core. One wave per (head, 16-query tile), 128-key blocks.
// D = 1024, Nk = 2048 hardcoded so all strides fold into immediates.
//   qp  : [Nq][D] bf16, pre-scaled by c^-0.5
//   kp  : [Nk][D] bf16 (row-major => contiguous B-frag for Q@K^T)
//   vpT : [D][Nk] bf16 (transposed => contiguous B-frag for P@V)
// Row sums of P via P @ ones on the matrix pipe; only row-max uses shuffles.
// ---------------------------------------------------------------------------
__global__ __launch_bounds__(32, 1) void attn_fmha(
    const __bf16* __restrict__ qp, const __bf16* __restrict__ kp,
    const __bf16* __restrict__ vpT, __bf16* __restrict__ xout) {
  constexpr int D = 1024;
  constexpr int NK = 2048;
  __shared__ __attribute__((aligned(32))) __bf16 Plds[16 * 128];

  const int head = blockIdx.y;
  const int q0 = blockIdx.x * 16;
  const int c0 = head * 64;
  const unsigned lane = threadIdx.x & 31u;
  const unsigned g = lane >> 4, nl = lane & 15;

  // Loop-invariant per-lane offsets.
  const unsigned offBD = nl * D + g * 16;   // B-frag lane offset, ld = D
  const unsigned offBN = nl * NK + g * 16;  // B-frag lane offset, ld = NK
  const unsigned offAD = nl * D + g * 16;   // A-frag lane offset, ld = D
  const unsigned offAP = nl * 128 + g * 16; // A-frag lane offset in Plds

  // All-ones B fragment for row-sum WMMA.
  v16bf ones;
#pragma unroll
  for (int e = 0; e < 16; ++e) ones[e] = (__bf16)1.0f;

  // Q A-fragments: key-loop invariant.
  const __bf16* qbase = qp + (size_t)q0 * D + c0;
  v16bf aq0 = frag_a_at(qbase + offAD);
  v16bf aq1 = frag_a_at(qbase + offAD + 32);

  float mrun[8], lrun[8];
  v8f o[4];
#pragma unroll
  for (int j = 0; j < 4; ++j) o[j] = (v8f){};
#pragma unroll
  for (int r = 0; r < 8; ++r) { mrun[r] = -__builtin_inff(); lrun[r] = 0.f; }

  const __bf16* vhead = vpT + (size_t)c0 * NK;

  for (int kb = 0; kb < NK; kb += 128) {
    const __bf16* kbase = kp + (size_t)kb * D + c0;  // uniform per block

    // ---- scores S[16 x 128] = Qtile @ K^T (8 n-tiles, K-dim 64, 2 steps) ---
    v8f s[8];
#pragma unroll
    for (int j = 0; j < 8; ++j) {
      v16bf b0 = frag_b_at(kbase + offBD + j * 16 * D);       // kk = 0
      s[j] = wmma_bf16(aq0, b0, (v8f){});
      v16bf b1 = frag_b_at(kbase + offBD + j * 16 * D + 32);  // kk = 32
      s[j] = wmma_bf16(aq1, b1, s[j]);
    }

    // ---- row max: local over frags, then off-major butterfly ----
    float bm[8];
#pragma unroll
    for (int r = 0; r < 8; ++r) {
      float v = s[0][r];
#pragma unroll
      for (int j = 1; j < 8; ++j) v = fmaxf(v, s[j][r]);
      bm[r] = v;
    }
#pragma unroll
    for (int off = 1; off < 16; off <<= 1) {
#pragma unroll
      for (int r = 0; r < 8; ++r) bm[r] = fmaxf(bm[r], __shfl_xor(bm[r], off, 32));
    }

    // ---- running max update, rescale O/l, exponentiate, stage P in LDS ----
    float alpha[8];
#pragma unroll
    for (int r = 0; r < 8; ++r) {
      float mnew = fmaxf(mrun[r], bm[r]);
      alpha[r] = __expf(mrun[r] - mnew);
      mrun[r] = mnew;
      lrun[r] *= alpha[r];
#pragma unroll
      for (int j = 0; j < 8; ++j) {
        float p = __expf(s[j][r] - mnew);
        Plds[(r + 8 * g) * 128 + 16 * j + nl] = (__bf16)p;
      }
#pragma unroll
      for (int j = 0; j < 4; ++j) o[j][r] *= alpha[r];
    }
    __syncthreads();  // P (C-frag layout) -> re-read as A-frags

    // ---- O += P @ V ; rowsum(P) += P @ ones (4 k-steps of 32) ----
    v8f psum = (v8f){};
#pragma unroll
    for (int kc = 0; kc < 4; ++kc) {
      v16bf pa = frag_a_at(&Plds[0] + offAP + 32 * kc);
      psum = wmma_bf16(pa, ones, psum);
      const __bf16* vk = vhead + kb + 32 * kc;  // uniform
#pragma unroll
      for (int j = 0; j < 4; ++j) {
        v16bf b = frag_b_at(vk + offBN + j * 16 * NK);
        o[j] = wmma_bf16(pa, b, o[j]);
      }
    }
#pragma unroll
    for (int r = 0; r < 8; ++r) lrun[r] += psum[r];
    __syncthreads();  // protect Plds before next block overwrites it
  }

  // ---- normalize and emit bf16 x (input to the Wo GEMM) ----
  __bf16* xbase = xout + (size_t)q0 * D + c0;
  const unsigned offX = (8 * g) * D + nl;
#pragma unroll
  for (int r = 0; r < 8; ++r) {
    float inv = 1.f / lrun[r];
#pragma unroll
    for (int j = 0; j < 4; ++j)
      xbase[offX + r * D + 16 * j] = (__bf16)(o[j][r] * inv);
  }
}

// ---------------------------------------------------------------------------
// Host-side orchestration
// ---------------------------------------------------------------------------
extern "C" void kernel_launch(void* const* d_in, const int* in_sizes, int n_in,
                              void* d_out, int out_size, void* d_ws, size_t ws_size,
                              hipStream_t stream) {
  const float* qin = (const float*)d_in[0];
  const float* kin = (const float*)d_in[1];
  const float* vin = (const float*)d_in[2];
  const float* Wq  = (const float*)d_in[3];
  const float* Wk  = (const float*)d_in[4];
  const float* Wv  = (const float*)d_in[5];
  const float* Wo  = (const float*)d_in[6];
  const float* bo  = (const float*)d_in[7];

  const int Nq = 2048, Nk = 2048, Dq = 1024, Dc = 768;

  char* ws = (char*)d_ws;
  size_t off = 0;
  auto wsalloc = [&](size_t bytes) {
    void* p = ws + off;
    off += (bytes + 255) & ~(size_t)255;
    return p;
  };
  __bf16* qb   = (__bf16*)wsalloc((size_t)Nq * Dq * 2);  // q   bf16 [Nq][Dq]
  __bf16* kb   = (__bf16*)wsalloc((size_t)Nk * Dc * 2);  // k   bf16 [Nk][Dc]
  __bf16* vb   = (__bf16*)wsalloc((size_t)Nk * Dc * 2);  // v   bf16 [Nk][Dc]
  __bf16* WqT  = (__bf16*)wsalloc((size_t)Dq * Dq * 2);  // Wq^T [Dq][Dq]
  __bf16* WkT  = (__bf16*)wsalloc((size_t)Dq * Dc * 2);  // Wk^T [Dq][Dc]
  __bf16* WvT  = (__bf16*)wsalloc((size_t)Dq * Dc * 2);  // Wv^T [Dq][Dc]
  __bf16* WoT  = (__bf16*)wsalloc((size_t)Dq * Dq * 2);  // Wo^T [Dq][Dq]
  __bf16* qpb  = (__bf16*)wsalloc((size_t)Nq * Dq * 2);  // (q@Wq)*scale [Nq][Dq]
  __bf16* kpb  = (__bf16*)wsalloc((size_t)Nk * Dq * 2);  // k@Wk [Nk][Dq]
  __bf16* vpT  = (__bf16*)wsalloc((size_t)Dq * Nk * 2);  // (v@Wv)^T [Dq][Nk]
  __bf16* xb   = (__bf16*)wsalloc((size_t)Nq * Dq * 2);  // attention out [Nq][Dq]
  (void)ws_size; (void)in_sizes; (void)n_in; (void)out_size;

  // --- convert inputs / transpose weights into bf16 ---
  cvt_f32_bf16<<<(Nq * Dq + 255) / 256, 256, 0, stream>>>(qin, qb, Nq * Dq);
  cvt_f32_bf16<<<(Nk * Dc + 255) / 256, 256, 0, stream>>>(kin, kb, Nk * Dc);
  cvt_f32_bf16<<<(Nk * Dc + 255) / 256, 256, 0, stream>>>(vin, vb, Nk * Dc);
  cvt_f32_bf16_T<<<dim3((Dq + 255) / 256, Dq), 256, 0, stream>>>(Wq, WqT, Dq, Dq);
  cvt_f32_bf16_T<<<dim3((Dq + 255) / 256, Dc), 256, 0, stream>>>(Wk, WkT, Dc, Dq);
  cvt_f32_bf16_T<<<dim3((Dq + 255) / 256, Dc), 256, 0, stream>>>(Wv, WvT, Dc, Dq);
  cvt_f32_bf16_T<<<dim3((Dq + 255) / 256, Dq), 256, 0, stream>>>(Wo, WoT, Dq, Dq);

  // --- projections (WMMA): qp scaled by c^-0.5 = 1/8 (head dim 64) ---
  gemm_bf16_wmma<OUT_BF16, 1024><<<dim3(Dq / 64, Nq / 64), 32, 0, stream>>>(
      qb, WqT, qpb, nullptr, Nq, Dq, 0.125f);
  gemm_bf16_wmma<OUT_BF16, 768><<<dim3(Dq / 64, Nk / 64), 32, 0, stream>>>(
      kb, WkT, kpb, nullptr, Nk, Dq, 1.0f);
  gemm_bf16_wmma<OUT_BF16_T, 768><<<dim3(Dq / 64, Nk / 64), 32, 0, stream>>>(
      vb, WvT, vpT, nullptr, Nk, Dq, 1.0f);

  // --- 16-head flash attention ---
  attn_fmha<<<dim3(Nq / 16, 16), 32, 0, stream>>>(qpb, kpb, vpT, xb);

  // --- output projection + bias, f32 out ---
  gemm_bf16_wmma<OUT_F32_BIAS, 1024><<<dim3(Dq / 64, Nq / 64), 32, 0, stream>>>(
      xb, WoT, (float*)d_out, bo, Nq, Dq, 1.0f);
}
